// MambaBlock_20074677141529
// MI455X (gfx1250) — matrix operand order repaired and checked by reading, not compiled, not run on hardware
//
#include <hip/hip_runtime.h>
#include <hip/hip_bf16.h>

// ---------------------------------------------------------------------------
// Mamba block forward for MI455X (gfx1250, wave32, WMMA)
// B=16, L=512, d_model=768, d_inner=1536, n_state=64, dt_rank=4, d_conv=4
// ---------------------------------------------------------------------------

typedef __bf16 bf16;
typedef __attribute__((ext_vector_type(8)))  __bf16 v8bf;
typedef __attribute__((ext_vector_type(16))) __bf16 v16bf;
typedef __attribute__((ext_vector_type(8)))  float  v8f;

#define BATCH   16
#define SEQ     512
#define TOK     (BATCH * SEQ)        // 8192
#define NE      768                  // n_embd
#define DI      1536                 // d_inner
#define NST     64                   // d_state
#define NXP     144                  // dt_rank + 2*n padded 132 -> 144

#define MBLK    128
#define NBLK    128
#define KSTEP   32
#define LDSK    40                   // padded LDS row stride (bf16 elems)
#define TCHUNK  8                    // scan timesteps staged per barrier

// ---------------------------------------------------------------------------
// Register staging for the double-buffered GEMM pipeline
// ---------------------------------------------------------------------------
struct Stage { v8bf a0, a1, b0, b1; };

__device__ __forceinline__ void stage_load(const bf16* __restrict__ A,
                                           const bf16* __restrict__ Bt,
                                           int N, int K, int m0, int n0, int k0,
                                           int sr, int sc, Stage& s) {
  const bf16* pa = A + (size_t)(m0 + sr) * K + (k0 + sc);
  s.a0 = *(const v8bf*)pa;
  s.a1 = *(const v8bf*)(pa + 8);
  const int gn = n0 + sr;
  if (gn < N) {
    const bf16* pb = Bt + (size_t)gn * K + (k0 + sc);
    s.b0 = *(const v8bf*)pb;
    s.b1 = *(const v8bf*)(pb + 8);
  } else {
#pragma unroll
    for (int e = 0; e < 8; ++e) { s.b0[e] = (bf16)0.0f; s.b1[e] = (bf16)0.0f; }
  }
}

__device__ __forceinline__ void stage_store(bf16 (*As)[LDSK], bf16 (*Bs)[LDSK],
                                            int sr, int sc, const Stage& s) {
  *(v8bf*)&As[sr][sc]     = s.a0;
  *(v8bf*)&As[sr][sc + 8] = s.a1;
  *(v8bf*)&Bs[sr][sc]     = s.b0;
  *(v8bf*)&Bs[sr][sc + 8] = s.b1;
}

// ---------------------------------------------------------------------------
// Generic bf16 WMMA GEMM:  C[M][N] (f32) = A[M][K] (bf16, row major)
//                                        x Bt[N][K] (bf16, B transposed)
// Block = 256 threads = 8 waves, block tile 128x128, wave tile 64Mx32N.
// Double-buffered LDS software pipeline: one barrier per K-step, global
// loads of tile k+1 overlap the 8 WMMAs of tile k.
// M multiple of 128, K multiple of 32; N guarded (zero-padded in LDS).
// ---------------------------------------------------------------------------
__global__ __launch_bounds__(256)
void gemm_bf16_wmma(const bf16* __restrict__ A, const bf16* __restrict__ Bt,
                    float* __restrict__ C, int M, int N, int K) {
  __shared__ __align__(16) bf16 As[2][MBLK][LDSK];
  __shared__ __align__(16) bf16 Bs[2][NBLK][LDSK];

  const int tid  = threadIdx.x;
  const int wid  = tid >> 5;
  const int lane = tid & 31;
  const int m0 = blockIdx.y * MBLK;
  const int n0 = blockIdx.x * NBLK;
  const int wm = (wid >> 2) * 64;   // wave M offset in block tile (0 or 64)
  const int wn = (wid & 3) * 32;    // wave N offset in block tile (0..96)

  const int lr    = lane & 15;      // row/col within 16
  const int khalf = lane >> 4;      // 0 or 1

  v8f acc[4][2];
#pragma unroll
  for (int mt = 0; mt < 4; ++mt)
#pragma unroll
    for (int nt = 0; nt < 2; ++nt)
#pragma unroll
      for (int e = 0; e < 8; ++e) acc[mt][nt][e] = 0.0f;

  // staging assignment: each thread stages one 16-bf16 (32B) chunk of A and B
  const int sr = tid >> 1;          // 0..127 : row of As / row (=n) of Bs
  const int sc = (tid & 1) * 16;    // 0 or 16

  const int nsteps = K / KSTEP;

  Stage st;
  stage_load(A, Bt, N, K, m0, n0, 0, sr, sc, st);
  stage_store(As[0], Bs[0], sr, sc, st);

  for (int step = 0; step < nsteps; ++step) {
    __syncthreads();                 // LDS buffer `cur` ready; `nxt` free
    const int  cur      = step & 1;
    const bool has_next = (step + 1) < nsteps;

    // issue global loads for the next tile while we compute this one
    if (has_next)
      stage_load(A, Bt, N, K, m0, n0, (step + 1) * KSTEP, sr, sc, st);
    if (step + 2 < nsteps) {         // warm L2 two tiles ahead
      __builtin_prefetch(A + (size_t)(m0 + sr) * K + ((step + 2) * KSTEP + sc), 0, 0);
      const int gn = n0 + sr;
      if (gn < N)
        __builtin_prefetch(Bt + (size_t)gn * K + ((step + 2) * KSTEP + sc), 0, 0);
    }

    // ---- load fragments per ISA VGPR layout ----
    // A 16x32: lane<16 -> K {0..7,16..23}; lane>=16 -> K {8..15,24..31}
    v16bf af[4];
#pragma unroll
    for (int mt = 0; mt < 4; ++mt) {
      const int row = wm + mt * 16 + lr;
      const int kb  = khalf * 8;
      v8bf lo = *(const v8bf*)&As[cur][row][kb];
      v8bf hi = *(const v8bf*)&As[cur][row][kb + 16];
      af[mt] = __builtin_shufflevector(lo, hi, 0, 1, 2, 3, 4, 5, 6, 7,
                                       8, 9, 10, 11, 12, 13, 14, 15);
    }
    // B 32x16: lane<16 -> col=lane, K {0..7,16..23} ; lane>=16 -> col=lane-16, K {8..15,24..31}
    v16bf bfr[2];
#pragma unroll
    for (int nt = 0; nt < 2; ++nt) {
      const int col = wn + nt * 16 + lr;
      const int kb  = khalf * 8;
      v8bf lo = *(const v8bf*)&Bs[cur][col][kb];
      v8bf hi = *(const v8bf*)&Bs[cur][col][kb + 16];
      bfr[nt] = __builtin_shufflevector(lo, hi, 0, 1, 2, 3, 4, 5, 6, 7,
                                        8, 9, 10, 11, 12, 13, 14, 15);
    }
#pragma unroll
    for (int mt = 0; mt < 4; ++mt)
#pragma unroll
      for (int nt = 0; nt < 2; ++nt)
        acc[mt][nt] = __builtin_amdgcn_wmma_f32_16x16x32_bf16(
            false, af[mt], false, bfr[nt], (short)0, acc[mt][nt], false, false);

    // park the next tile into the ping-pong buffer (read after next barrier)
    if (has_next)
      stage_store(As[cur ^ 1], Bs[cur ^ 1], sr, sc, st);
  }

  // ---- store C: lane half selects M 0..7 / 8..15 ; n = lane&15 ----
#pragma unroll
  for (int mt = 0; mt < 4; ++mt)
#pragma unroll
    for (int nt = 0; nt < 2; ++nt) {
      const int gmBase = m0 + wm + mt * 16 + khalf * 8;
      const int gn     = n0 + wn + nt * 16 + lr;
      if (gn < N) {
#pragma unroll
        for (int r = 0; r < 8; ++r)
          C[(size_t)(gmBase + r) * N + gn] = acc[mt][nt][r];
      }
    }
}

// ---------------------------------------------------------------------------
// Weight transpose + f32->bf16:  W[K][N] -> Wt[Npad][K], zero pad n >= N
// ---------------------------------------------------------------------------
__global__ void transpose_cvt(const float* __restrict__ W, bf16* __restrict__ Wt,
                              int K, int N, int Npad) {
  const long long idx = (long long)blockIdx.x * blockDim.x + threadIdx.x;
  if (idx >= (long long)Npad * K) return;
  const int n = (int)(idx % Npad);
  const int k = (int)(idx / Npad);
  Wt[(size_t)n * K + k] = (n < N) ? (bf16)W[(size_t)k * N + n] : (bf16)0.0f;
}

// ---------------------------------------------------------------------------
// f32 -> bf16 elementwise
// ---------------------------------------------------------------------------
__global__ void cvt_bf16(const float* __restrict__ src, bf16* __restrict__ dst,
                         long long count) {
  const long long idx = (long long)blockIdx.x * blockDim.x + threadIdx.x;
  if (idx < count) dst[idx] = (bf16)src[idx];
}

// ---------------------------------------------------------------------------
// Causal depthwise conv (4 taps) + bias + SiLU, over x_ssm half of xr.
// xr: [TOK][2*DI], x_ssm = cols 0..DI-1.  Outputs f32 and bf16.
// ---------------------------------------------------------------------------
__global__ void conv_silu(const float* __restrict__ xr,
                          const float* __restrict__ conv_w,
                          const float* __restrict__ conv_b,
                          float* __restrict__ xs_f32, bf16* __restrict__ xs_bf) {
  const long long idx = (long long)blockIdx.x * blockDim.x + threadIdx.x;
  if (idx >= (long long)TOK * DI) return;
  const int d     = (int)(idx % DI);
  const int token = (int)(idx / DI);
  const int t     = token % SEQ;
  float acc = conv_b[d];
#pragma unroll
  for (int j = 0; j < 4; ++j) {
    const int tt = t - 3 + j;
    if (tt >= 0)
      acc += xr[(size_t)(token - 3 + j) * (2 * DI) + d] * conv_w[d * 4 + j];
  }
  const float s = acc / (1.0f + __expf(-acc));   // SiLU
  xs_f32[idx] = s;
  xs_bf[idx]  = (bf16)s;
}

// ---------------------------------------------------------------------------
// Fused selective scan. One thread per (batch, channel); 64-state h[] and A
// row fully in registers. B/C/dlt rows staged to LDS in chunks of TCHUNK
// timesteps (one barrier pair per chunk, fully coalesced staging loads).
// Fuses softplus(delta), recurrence, D-skip, SiLU(res) gate, bf16 emit.
// grid = (DI/128, BATCH), block = 128
// ---------------------------------------------------------------------------
__global__ __launch_bounds__(128)
void scan_kernel(const float* __restrict__ xdbl,   // [TOK][NXP]
                 const float* __restrict__ xs,     // [TOK][DI]
                 const float* __restrict__ xr,     // [TOK][2*DI] (res half)
                 const float* __restrict__ W_dt,   // [4][DI]
                 const float* __restrict__ b_dt,   // [DI]
                 const float* __restrict__ A_log,  // [DI][64]
                 const float* __restrict__ Dp,     // [DI]
                 bf16* __restrict__ z)              // [TOK][DI]
{
  const int tid = threadIdx.x;
  const int d   = blockIdx.x * 128 + tid;
  const int b   = blockIdx.y;

  __shared__ float sBuf[TCHUNK][NXP];   // dlt[0..3], B[4..67], C[68..131]

  float a[NST], h[NST];
#pragma unroll
  for (int n = 0; n < NST; ++n) {
    a[n] = -__expf(A_log[(size_t)d * NST + n]);
    h[n] = 0.0f;
  }
  const float w0 = W_dt[0 * DI + d], w1 = W_dt[1 * DI + d];
  const float w2 = W_dt[2 * DI + d], w3 = W_dt[3 * DI + d];
  const float bd = b_dt[d];
  const float Dd = Dp[d];

  for (int t0 = 0; t0 < SEQ; t0 += TCHUNK) {
    __syncthreads();
    // stage TCHUNK rows of x_dbl (coalesced: consecutive tid -> consecutive col)
    const float* src = xdbl + ((size_t)b * SEQ + t0) * NXP;
#pragma unroll
    for (int i = tid; i < TCHUNK * NXP; i += 128)
      ((float*)sBuf)[i] = src[i];
    __syncthreads();

    for (int ts = 0; ts < TCHUNK; ++ts) {
      const size_t token = (size_t)b * SEQ + (t0 + ts);
      const float* row = sBuf[ts];

      const float pre = row[0] * w0 + row[1] * w1 + row[2] * w2 + row[3] * w3 + bd;
      const float delta = (pre > 20.0f) ? pre : __logf(1.0f + __expf(pre)); // softplus
      const float xv = xs[token * DI + d];
      const float dx = delta * xv;

      float y = 0.0f;
#pragma unroll
      for (int n = 0; n < NST; ++n) {
        const float dA = __expf(delta * a[n]);
        h[n] = dA * h[n] + dx * row[4 + n];
        y += row[4 + NST + n] * h[n];
      }
      const float r = xr[token * (2 * DI) + DI + d];
      const float g = r / (1.0f + __expf(-r));     // SiLU(res)
      z[token * DI + d] = (bf16)((y + Dd * xv) * g);
    }
  }
}

// ---------------------------------------------------------------------------
extern "C" void kernel_launch(void* const* d_in, const int* in_sizes, int n_in,
                              void* d_out, int out_size, void* d_ws, size_t ws_size,
                              hipStream_t stream) {
  (void)in_sizes; (void)n_in; (void)out_size; (void)ws_size;
  const float* x      = (const float*)d_in[0];
  const float* W_in   = (const float*)d_in[1];
  const float* conv_w = (const float*)d_in[2];
  const float* conv_b = (const float*)d_in[3];
  const float* W_x    = (const float*)d_in[4];
  const float* W_dt   = (const float*)d_in[5];
  const float* b_dt   = (const float*)d_in[6];
  const float* A_log  = (const float*)d_in[7];
  const float* Dp     = (const float*)d_in[8];
  const float* W_out  = (const float*)d_in[9];
  float* out = (float*)d_out;

  // ---- workspace carving (256B aligned) ----
  char* ws = (char*)d_ws;
  size_t off = 0;
  auto alloc = [&](size_t bytes) -> void* {
    void* p = (void*)(ws + off);
    off += (bytes + 255) & ~(size_t)255;
    return p;
  };
  bf16*  xbf   = (bf16*) alloc((size_t)TOK * NE * 2);        // x in bf16
  bf16*  WinT  = (bf16*) alloc((size_t)(2 * DI) * NE * 2);   // [3072][768]
  bf16*  WxT   = (bf16*) alloc((size_t)NXP * DI * 2);        // [144][1536]
  bf16*  WoutT = (bf16*) alloc((size_t)NE * DI * 2);         // [768][1536]
  float* xr    = (float*)alloc((size_t)TOK * (2 * DI) * 4);  // in-proj result
  float* xsf   = (float*)alloc((size_t)TOK * DI * 4);        // conv+silu f32
  bf16*  xsbf  = (bf16*) alloc((size_t)TOK * DI * 2);        // conv+silu bf16
  float* xdbl  = (float*)alloc((size_t)TOK * NXP * 4);       // x_dbl (padded)
  bf16*  zbf   = (bf16*) alloc((size_t)TOK * DI * 2);        // gated scan out

  // ---- prep: weight transpose/convert + x convert ----
  {
    long long cnt = (long long)(2 * DI) * NE;
    transpose_cvt<<<dim3((unsigned)((cnt + 255) / 256)), 256, 0, stream>>>(
        W_in, WinT, NE, 2 * DI, 2 * DI);
  }
  {
    long long cnt = (long long)NXP * DI;
    transpose_cvt<<<dim3((unsigned)((cnt + 255) / 256)), 256, 0, stream>>>(
        W_x, WxT, DI, 132, NXP);
  }
  {
    long long cnt = (long long)NE * DI;
    transpose_cvt<<<dim3((unsigned)((cnt + 255) / 256)), 256, 0, stream>>>(
        W_out, WoutT, DI, NE, NE);
  }
  {
    long long cnt = (long long)TOK * NE;
    cvt_bf16<<<dim3((unsigned)((cnt + 255) / 256)), 256, 0, stream>>>(x, xbf, cnt);
  }

  // ---- GEMM 1: xr = x @ W_in   [8192 x 3072 x 768] ----
  gemm_bf16_wmma<<<dim3((2 * DI + NBLK - 1) / NBLK, TOK / MBLK), 256, 0, stream>>>(
      xbf, WinT, xr, TOK, 2 * DI, NE);

  // ---- conv + bias + SiLU ----
  {
    long long cnt = (long long)TOK * DI;
    conv_silu<<<dim3((unsigned)((cnt + 255) / 256)), 256, 0, stream>>>(
        xr, conv_w, conv_b, xsf, xsbf);
  }

  // ---- GEMM 2: x_dbl = x_ssm @ W_x   [8192 x 144 x 1536] ----
  gemm_bf16_wmma<<<dim3((NXP + NBLK - 1) / NBLK, TOK / MBLK), 256, 0, stream>>>(
      xsbf, WxT, xdbl, TOK, NXP, DI);

  // ---- fused selective scan + gate ----
  scan_kernel<<<dim3(DI / 128, BATCH), 128, 0, stream>>>(
      xdbl, xsf, xr, W_dt, b_dt, A_log, Dp, zbf);

  // ---- GEMM 3: out = z @ W_out   [8192 x 768 x 1536] ----
  gemm_bf16_wmma<<<dim3((NE + NBLK - 1) / NBLK, TOK / MBLK), 256, 0, stream>>>(
      zbf, WoutT, out, TOK, NE, DI);
}
